// SparseAutoencoder_39135742001983
// MI455X (gfx1250) — matrix operand order repaired and checked by reading, not compiled, not run on hardware
//
#include <hip/hip_runtime.h>

typedef unsigned int  uint;
typedef unsigned short ushort;
typedef __attribute__((ext_vector_type(16))) __bf16 v16bf;
typedef __attribute__((ext_vector_type(8)))  float  v8f;

union FragU { v16bf v; uint u[8]; };

#define D_MODEL  4096
#define D_HIDDEN 32768
#define BATCH    2048
#define TOPK     128
#define NDEAD    2048

__device__ __forceinline__ ushort f2bf(float f) {
  uint u = __float_as_uint(f);
  u += 0x7FFFu + ((u >> 16) & 1u);            // round-to-nearest-even
  return (ushort)(u >> 16);
}

// monotonic float -> uint mapping (ascending)
__device__ __forceinline__ uint mapf(float f) {
  uint u = __float_as_uint(f);
  return (u & 0x80000000u) ? ~u : (u | 0x80000000u);
}

// ---- CDNA5 async global->LDS copy (ASYNCcnt-tracked, no VGPR round trip) ----
__device__ __forceinline__ void async_g2l_b128(uint lds_off, const ushort* gptr) {
  asm volatile("global_load_async_to_lds_b128 %0, %1, off"
               :: "v"(lds_off), "v"(gptr) : "memory");
}
__device__ __forceinline__ void wait_async0() {
#if __has_builtin(__builtin_amdgcn_s_wait_asynccnt)
  __builtin_amdgcn_s_wait_asynccnt(0);
#else
  asm volatile("s_wait_asynccnt 0x0" ::: "memory");
#endif
}

// ---------------------------------------------------------------------------
// 1) LayerNorm (ddof=1) + subtract b_pre, emit bf16 activations + mu/std
// ---------------------------------------------------------------------------
__global__ __launch_bounds__(256)
void ln_kernel(const float* __restrict__ x, const float* __restrict__ b_pre,
               ushort* __restrict__ xc, float* __restrict__ mu, float* __restrict__ sd) {
  const int row = blockIdx.x, tid = threadIdx.x;
  const float* xr = x + (size_t)row * D_MODEL;
  float s = 0.f, s2 = 0.f;
  for (int i = tid; i < D_MODEL; i += 256) { float v = xr[i]; s += v; s2 += v * v; }
  __shared__ float rs[256], rq[256];
  rs[tid] = s; rq[tid] = s2; __syncthreads();
  for (int off = 128; off > 0; off >>= 1) {
    if (tid < off) { rs[tid] += rs[tid + off]; rq[tid] += rq[tid + off]; }
    __syncthreads();
  }
  const float mean = rs[0] * (1.0f / D_MODEL);
  const float var  = (rq[0] - (float)D_MODEL * mean * mean) * (1.0f / (D_MODEL - 1));
  const float st   = sqrtf(fmaxf(var, 0.f));
  if (tid == 0) { mu[row] = mean; sd[row] = st; }
  const float inv = 1.0f / (st + 1e-5f);
  for (int i = tid; i < D_MODEL; i += 256)
    xc[(size_t)row * D_MODEL + i] = f2bf((xr[i] - mean) * inv - b_pre[i]);
}

// ---------------------------------------------------------------------------
// 2) Tiled bf16 WMMA GEMM: C[M,N] = A_bf16[M,K] * B_f32[K,N] (+bias[, *scale+shift])
//    128x128 block tile, BK=32, 8 waves x (4x2) 16x16 WMMA tiles.
//    Double-buffered LDS; A staged with GLOBAL_LOAD_ASYNC_TO_LDS_B128,
//    B staged through VGPRs with inline f32->bf16 convert.
//    GATHER=1: B rows indirected through bidx, epilogue applies *scale[m]+shift[m].
// ---------------------------------------------------------------------------
template <int GATHER>
__global__ __launch_bounds__(256)
void gemm_bf16_wmma(const ushort* __restrict__ A, int lda,
                    const float*  __restrict__ B, int ldb,
                    const int*    __restrict__ bidx,
                    float* __restrict__ C, int ldc,
                    int M, int N, int Ktot,
                    const float* __restrict__ bias,
                    const float* __restrict__ scale,
                    const float* __restrict__ shift) {
  __shared__ __align__(16) ushort As[2][128 * 40];   // [m][k], padded stride 40
  __shared__ __align__(16) ushort Bs[2][128 * 40];   // [n][k], padded stride 40

  const int tid  = threadIdx.x;
  const int lane = tid & 31;
  const int wid  = tid >> 5;
  const int wm   = wid & 1;            // 2 wave rows  (64 rows each)
  const int wn   = wid >> 1;           // 4 wave cols  (32 cols each)
  const int m0   = blockIdx.x * 128;   // block-M fastest -> B panel shared in L2
  const int n0   = blockIdx.y * 128;

  v8f acc[4][2] = {};

  const int nl  = lane & 15;
  const int akb = (lane < 16) ? 0 : 16;   // A: contiguous 16-K per half-wave
  const int bkb = (lane < 16) ? 0 : 16;   // B: contiguous 16-K per half-wave

  // per-thread staging coordinates
  const int ar0 = (tid + 0)   >> 2, ac0 = ((tid + 0)   & 3) << 3;
  const int ar1 = (tid + 256) >> 2, ac1 = ((tid + 256) & 3) << 3;

  float4 breg[4];

  // ---- prologue: stage tile 0 into buffer 0 ----
  async_g2l_b128((uint)(size_t)&As[0][ar0 * 40 + ac0], A + (size_t)(m0 + ar0) * lda + ac0);
  async_g2l_b128((uint)(size_t)&As[0][ar1 * 40 + ac1], A + (size_t)(m0 + ar1) * lda + ac1);
#pragma unroll
  for (int i = 0; i < 4; ++i) {
    const int v = tid + i * 256;
    const int kr = v >> 5, n4 = (v & 31) << 2;
    const int br = GATHER ? bidx[kr] : kr;
    breg[i] = *(const float4*)(B + (size_t)br * ldb + n0 + n4);
  }
#pragma unroll
  for (int i = 0; i < 4; ++i) {
    const int v = tid + i * 256;
    const int kr = v >> 5, n4 = (v & 31) << 2;
    Bs[0][(n4 + 0) * 40 + kr] = f2bf(breg[i].x);
    Bs[0][(n4 + 1) * 40 + kr] = f2bf(breg[i].y);
    Bs[0][(n4 + 2) * 40 + kr] = f2bf(breg[i].z);
    Bs[0][(n4 + 3) * 40 + kr] = f2bf(breg[i].w);
  }
  wait_async0();
  __syncthreads();

  int cur = 0;
  for (int k0 = 0; k0 < Ktot; k0 += 32) {
    const int nxt = cur ^ 1;
    const bool more = (k0 + 32) < Ktot;
    const int kk = k0 + 32;

    // ---- stage tile t+1 while computing tile t ----
    if (more) {
      async_g2l_b128((uint)(size_t)&As[nxt][ar0 * 40 + ac0],
                     A + (size_t)(m0 + ar0) * lda + kk + ac0);
      async_g2l_b128((uint)(size_t)&As[nxt][ar1 * 40 + ac1],
                     A + (size_t)(m0 + ar1) * lda + kk + ac1);
#pragma unroll
      for (int i = 0; i < 4; ++i) {
        const int v = tid + i * 256;
        const int kr = v >> 5, n4 = (v & 31) << 2;
        const int br = GATHER ? bidx[kk + kr] : (kk + kr);
        breg[i] = *(const float4*)(B + (size_t)br * ldb + n0 + n4);
      }
      if (!GATHER)
        __builtin_prefetch(B + (size_t)(kk + 32 + (tid >> 5)) * ldb + n0 + ((tid & 31) << 2), 0, 1);
    }

    // ---- compute tile t ----
    FragU bf[2];
#pragma unroll
    for (int nt = 0; nt < 2; ++nt) {
      const int nb = (wn * 32 + nt * 16 + nl) * 40 + bkb;
#pragma unroll
      for (int v = 0; v < 8; ++v) bf[nt].u[v] = *(const uint*)&Bs[cur][nb + 2 * v];
    }
#pragma unroll
    for (int mt = 0; mt < 4; ++mt) {
      FragU af;
      const int ab = (wm * 64 + mt * 16 + nl) * 40;
#pragma unroll
      for (int v = 0; v < 4; ++v) af.u[v]     = *(const uint*)&As[cur][ab + akb + 2 * v];
#pragma unroll
      for (int v = 0; v < 4; ++v) af.u[4 + v] = *(const uint*)&As[cur][ab + 8 + akb + 2 * v];
#pragma unroll
      for (int nt = 0; nt < 2; ++nt)
        acc[mt][nt] = __builtin_amdgcn_wmma_f32_16x16x32_bf16(
            false, af.v, false, bf[nt].v, (short)0, acc[mt][nt], false, false);
    }

    // ---- finish staging tile t+1 ----
    if (more) {
#pragma unroll
      for (int i = 0; i < 4; ++i) {
        const int v = tid + i * 256;
        const int kr = v >> 5, n4 = (v & 31) << 2;
        Bs[nxt][(n4 + 0) * 40 + kr] = f2bf(breg[i].x);
        Bs[nxt][(n4 + 1) * 40 + kr] = f2bf(breg[i].y);
        Bs[nxt][(n4 + 2) * 40 + kr] = f2bf(breg[i].z);
        Bs[nxt][(n4 + 3) * 40 + kr] = f2bf(breg[i].w);
      }
      wait_async0();
    }
    __syncthreads();
    cur = nxt;
  }

  // ---- epilogue: + bias[n], optional *scale[m] + shift[m] ----
#pragma unroll
  for (int mt = 0; mt < 4; ++mt)
#pragma unroll
    for (int nt = 0; nt < 2; ++nt) {
      const int n = n0 + wn * 32 + nt * 16 + nl;
      const float bv = bias[n];
#pragma unroll
      for (int r = 0; r < 8; ++r) {
        const int m = m0 + wm * 64 + mt * 16 + r + ((lane >> 4) << 3);
        float c = acc[mt][nt][r] + bv;
        if (GATHER) c = c * scale[m] + shift[m];
        C[(size_t)m * ldc + n] = c;
      }
    }
}

// ---------------------------------------------------------------------------
// 3) Per-row top-128 (4-pass radix select) + fused sparse decode + un-normalize
// ---------------------------------------------------------------------------
__global__ __launch_bounds__(256)
void topk_recon(const float* __restrict__ pre, const float* __restrict__ wdec,
                const float* __restrict__ b_pre, const float* __restrict__ mu,
                const float* __restrict__ sd, float* __restrict__ out) {
  const int row = blockIdx.x, tid = threadIdx.x;
  const float* pr = pre + (size_t)row * D_HIDDEN;

  __shared__ int hist[256];
  __shared__ int sh_bin, sh_want;
  __shared__ float selv[TOPK];
  __shared__ int   seli[TOPK];
  __shared__ int nsel, tiec;

  uint prefix = 0, maskHi = 0;
  int want = TOPK;
  for (int pass = 0; pass < 4; ++pass) {
    const int shift = 24 - 8 * pass;
    hist[tid] = 0;
    if (tid == 0) { nsel = 0; tiec = 0; }
    __syncthreads();
    for (int i = tid; i < D_HIDDEN; i += 256) {
      uint u = mapf(pr[i]);
      if ((u & maskHi) == prefix) atomicAdd(&hist[(u >> shift) & 255], 1);
    }
    __syncthreads();
    if (tid == 0) {
      int cum = 0, b = 255;
      for (; b > 0; --b) { int c = hist[b]; if (cum + c >= want) break; cum += c; }
      sh_bin = b; sh_want = want - cum;
    }
    __syncthreads();
    prefix |= ((uint)sh_bin) << shift;
    maskHi |= 0xFFu << shift;
    want = sh_want;
    __syncthreads();
  }
  const uint thr = prefix;

  for (int i = tid; i < D_HIDDEN; i += 256) {
    float v = pr[i];
    uint u = mapf(v);
    if (u > thr) {
      int p = atomicAdd(&nsel, 1);
      selv[p] = fmaxf(v, 0.f); seli[p] = i;
    } else if (u == thr) {
      int t = atomicAdd(&tiec, 1);
      if (t < want) { int p = atomicAdd(&nsel, 1); selv[p] = fmaxf(v, 0.f); seli[p] = i; }
    }
  }
  __syncthreads();
  const int ns = nsel;

  float acc[16];
#pragma unroll
  for (int j = 0; j < 16; ++j) acc[j] = b_pre[tid + j * 256];
  for (int s = 0; s < ns; ++s) {
    const float v = selv[s];
    const float* wr = wdec + (size_t)seli[s] * D_MODEL;
#pragma unroll
    for (int j = 0; j < 16; ++j) acc[j] += v * wr[tid + j * 256];
  }
  const float st = sd[row], m = mu[row];
#pragma unroll
  for (int j = 0; j < 16; ++j)
    out[(size_t)row * D_MODEL + tid + j * 256] = acc[j] * st + m;
}

// ---------------------------------------------------------------------------
// 4) Dead-latent index compaction (deterministic, ascending order)
// ---------------------------------------------------------------------------
__global__ __launch_bounds__(256)
void dead_compact(const unsigned char* __restrict__ mask, int* __restrict__ idx) {
  __shared__ int cs[256];
  const int tid = threadIdx.x, base = tid * (D_HIDDEN / 256);
  int c = 0;
  for (int i = 0; i < D_HIDDEN / 256; ++i) c += mask[base + i] ? 1 : 0;
  cs[tid] = c; __syncthreads();
  if (tid == 0) { int a = 0; for (int i = 0; i < 256; ++i) { int t = cs[i]; cs[i] = a; a += t; } }
  __syncthreads();
  int p = cs[tid];
  for (int i = 0; i < D_HIDDEN / 256; ++i)
    if (mask[base + i]) idx[p++] = base + i;
}

// 5) Gather relu(pre_acts[:, dead]) into dense bf16 [BATCH x NDEAD]
__global__ __launch_bounds__(256)
void gather_dead(const float* __restrict__ pre, const int* __restrict__ didx,
                 ushort* __restrict__ Ad) {
  const int j = blockIdx.y * 256 + threadIdx.x;   // dead column 0..2047
  const int row = blockIdx.x;
  const float v = pre[(size_t)row * D_HIDDEN + didx[j]];
  Ad[(size_t)row * NDEAD + j] = f2bf(fmaxf(v, 0.f));
}

__global__ void write_meta(const int* __restrict__ nd, float* __restrict__ o) {
  if (threadIdx.x == 0) o[0] = (float)nd[0];
}

// ---------------------------------------------------------------------------
extern "C" void kernel_launch(void* const* d_in, const int* in_sizes, int n_in,
                              void* d_out, int out_size, void* d_ws, size_t ws_size,
                              hipStream_t stream) {
  const float* x      = (const float*)d_in[0];
  const float* w_enc  = (const float*)d_in[1];
  const float* w_dec  = (const float*)d_in[2];
  const float* b_enc  = (const float*)d_in[3];
  const float* b_pre  = (const float*)d_in[4];
  const unsigned char* dead_mask = (const unsigned char*)d_in[5];
  const int*   num_dead = (const int*)d_in[6];
  float* out = (float*)d_out;

  char* ws = (char*)d_ws;
  float*  mu   = (float*)(ws);
  float*  sd   = (float*)(ws + 8192);
  int*    didx = (int*)  (ws + 16384);
  ushort* xc   = (ushort*)(ws + 65536);                                  // 16 MB
  ushort* Ad   = (ushort*)(ws + 65536 + ((size_t)16 << 20));             //  8 MB
  float*  pre  = (float*) (ws + 65536 + ((size_t)24 << 20));             // 256 MB

  // 1) layernorm + bf16 pack
  ln_kernel<<<BATCH, 256, 0, stream>>>(x, b_pre, xc, mu, sd);

  // 2) encoder GEMM: pre = xc @ w_enc + b_enc   [2048 x 32768], K=4096
  gemm_bf16_wmma<0><<<dim3(BATCH / 128, D_HIDDEN / 128), 256, 0, stream>>>(
      xc, D_MODEL, w_enc, D_HIDDEN, nullptr,
      pre, D_HIDDEN, BATCH, D_HIDDEN, D_MODEL, b_enc, nullptr, nullptr);

  // 3) top-128 select + sparse decode + un-normalize -> recons
  topk_recon<<<BATCH, 256, 0, stream>>>(pre, w_dec, b_pre, mu, sd, out);

  // 4) auxk path: all dead latents selected (k_aux == num_dead)
  dead_compact<<<1, 256, 0, stream>>>(dead_mask, didx);
  gather_dead<<<dim3(BATCH, NDEAD / 256), 256, 0, stream>>>(pre, didx, Ad);
  gemm_bf16_wmma<1><<<dim3(BATCH / 128, D_MODEL / 128), 256, 0, stream>>>(
      Ad, NDEAD, w_dec, D_MODEL, didx,
      out + (size_t)BATCH * D_MODEL, D_MODEL, BATCH, D_MODEL, NDEAD,
      b_pre, sd, mu);

  // 5) scalar third output
  write_meta<<<1, 1, 0, stream>>>(num_dead, out + (size_t)2 * BATCH * D_MODEL);
}
